// VarianceAdaptor_25237227831799
// MI455X (gfx1250) — matrix operand order repaired and checked by reading, not compiled, not run on hardware
//
#include <hip/hip_runtime.h>
#include <hip/hip_bf16.h>
#include <math.h>

// ---- problem constants (match reference) ----
#define BB    32
#define SS    512
#define HH    256
#define FSS   256
#define TMAXX 2048
#define NB    256

typedef __attribute__((ext_vector_type(16))) __bf16 v16bf;
typedef __attribute__((ext_vector_type(8)))  float  v8f;

__device__ __forceinline__ unsigned short f2bf(float f) {
    unsigned u = __float_as_uint(f);
    if ((u & 0x7F800000u) == 0x7F800000u) return (unsigned short)(u >> 16); // inf/nan
    unsigned r = (u + 0x7FFFu + ((u >> 16) & 1u)) >> 16;
    return (unsigned short)r;
}

// ---------------------------------------------------------------------------
// Pack conv weight [3][256][256] f32 -> bf16 B-fragments in per-lane register
// image. Fragment fi = (k*16 + ntile)*8 + kc ; lane L holds col n = ntile*16 +
// (L&15); 16 K values kk = kc*32 + (L>>4)*8 + i (i<8), kc*32 + 16 + (L>>4)*8 + (i-8) (i>=8).  32B per lane.
// ---------------------------------------------------------------------------
__global__ void pack_conv_w(const float* __restrict__ w, unsigned short* __restrict__ dst) {
    int t = blockIdx.x * blockDim.x + threadIdx.x;
    if (t >= 384 * 32) return;
    int fi = t >> 5, L = t & 31;
    int k = fi / 128, rem = fi % 128, ntile = rem >> 3, kc = rem & 7;
    int n = ntile * 16 + (L & 15), grp = L >> 4;
    unsigned short* o = dst + (size_t)fi * 512 + L * 16;
#pragma unroll
    for (int i = 0; i < 16; ++i) {
        int kk = kc * 32 + (i < 8 ? grp * 8 + i : 16 + grp * 8 + (i - 8));
        o[i] = f2bf(w[((size_t)k * 256 + kk) * 256 + n]);
    }
}

// Pack linear weight [256][dim] f32 -> bf16 fragments, N padded to 16.
__global__ void pack_lin_w(const float* __restrict__ wl, int dim, unsigned short* __restrict__ dst) {
    int t = threadIdx.x;              // 256 threads = 8 frags * 32 lanes
    int fi = t >> 5, L = t & 31;
    int n = L & 15, grp = L >> 4;
    unsigned short* o = dst + (size_t)fi * 512 + L * 16;
#pragma unroll
    for (int i = 0; i < 16; ++i) {
        int kk = fi * 32 + (i < 8 ? grp * 8 + i : 16 + grp * 8 + (i - 8));
        o[i] = (n < dim) ? f2bf(wl[(size_t)kk * dim + n]) : (unsigned short)0;
    }
}

// ---------------------------------------------------------------------------
// Fused Conv1d(K=3,SAME) + bias + ReLU + LayerNorm for a 16-row tile.
// Block = 512 threads = 16 waves; wave w owns output cols [16w,16w+16).
// A-tile (18 rows x 256 bf16) staged in LDS: 16 main rows via async-to-LDS,
// 2 halo rows (zero padded at batch edges) via ds stores.
// ---------------------------------------------------------------------------
__global__ __launch_bounds__(512)
void conv_ln(const unsigned short* __restrict__ in,   // bf16 [B*S][256]
             const unsigned short* __restrict__ wp,   // packed bf16 fragments
             const float* __restrict__ bias,
             const float* __restrict__ gamma,
             const float* __restrict__ beta,
             unsigned short* __restrict__ out) {      // bf16 [B*S][256]
    __shared__ unsigned short als[18 * 256];
    __shared__ float psum[16 * 16], psq[16 * 16], mrow[16], rrow[16];

    int tile = blockIdx.x;                 // 1024 tiles
    int s0 = (tile * 16) & (SS - 1);
    long row0 = (long)tile * 16;
    int tid = threadIdx.x, wid = tid >> 5, L = tid & 31;

    // --- stage 16 main rows with async global->LDS copies (one b128/thread)
    {
        int r = tid >> 5;                  // 0..15
        int ch = tid & 31;                 // 16B chunk 0..31
        const unsigned short* gp = in + (row0 + r) * 256 + ch * 8;
        unsigned lo = (unsigned)(unsigned long long)(&als[(r + 1) * 256 + ch * 8]);
        asm volatile("global_load_async_to_lds_b128 %0, %1, off"
                     :: "v"(lo), "v"((unsigned long long)gp) : "memory");
    }
    // --- halo rows (s0-1 and s0+16), zero outside batch
    if (tid < 64) {
        int which = tid >> 5;              // 0 -> rowLocal 0, 1 -> rowLocal 17
        int ch = tid & 31;
        int sl = which ? (s0 + 16) : (s0 - 1);
        uint4 v = make_uint4(0u, 0u, 0u, 0u);
        if (sl >= 0 && sl < SS) {
            long rr = row0 - s0 + sl;
            v = *(const uint4*)(in + rr * 256 + ch * 8);
        }
        *(uint4*)(&als[(which ? 17 : 0) * 256 + ch * 8]) = v;
    }
    asm volatile("s_wait_asynccnt 0" ::: "memory");
    __syncthreads();

    int n = wid * 16 + (L & 15);
    int grp = L >> 4;
    float bcol = bias[n];
    v8f acc;
#pragma unroll
    for (int j = 0; j < 8; ++j) acc[j] = bcol;

#pragma unroll
    for (int k = 0; k < 3; ++k) {
#pragma unroll
        for (int kc = 0; kc < 8; ++kc) {
            union { v16bf v; uint4 q[2]; } A, Bf;
            int rowL = (L & 15) + k;
            int c0 = kc * 32;
            A.q[0] = *(const uint4*)(&als[rowL * 256 + c0 + grp * 8]);
            A.q[1] = *(const uint4*)(&als[rowL * 256 + c0 + 16 + grp * 8]);
            const unsigned short* bp = wp + ((size_t)((k * 16 + wid) * 8 + kc)) * 512 + L * 16;
            Bf.q[0] = *(const uint4*)(bp);
            Bf.q[1] = *(const uint4*)(bp + 8);
            acc = __builtin_amdgcn_wmma_f32_16x16x32_bf16(
                false, A.v, false, Bf.v, (short)0, acc, false, false);
        }
    }

    // ReLU then LayerNorm over the 256 cols of each row
    float rv[8];
#pragma unroll
    for (int j = 0; j < 8; ++j) rv[j] = fmaxf(acc[j], 0.f);
#pragma unroll
    for (int j = 0; j < 8; ++j) {
        float s = rv[j], q = rv[j] * rv[j];
#pragma unroll
        for (int m = 1; m < 16; m <<= 1) {
            s += __shfl_xor(s, m, 16);
            q += __shfl_xor(q, m, 16);
        }
        if ((L & 15) == j) {
            psum[wid * 16 + j + 8 * grp] = s;
            psq[wid * 16 + j + 8 * grp] = q;
        }
    }
    __syncthreads();
    if (tid < 16) {
        float s = 0.f, q = 0.f;
        for (int w = 0; w < 16; ++w) { s += psum[w * 16 + tid]; q += psq[w * 16 + tid]; }
        float m = s * (1.f / 256.f);
        float v = q * (1.f / 256.f) - m * m;
        mrow[tid] = m;
        rrow[tid] = rsqrtf(v + 1e-5f);
    }
    __syncthreads();
    float gg = gamma[n], bb = beta[n];
#pragma unroll
    for (int j = 0; j < 8; ++j) {
        int rr = j + 8 * grp;
        float y = (rv[j] - mrow[rr]) * rrow[rr] * gg + bb;
        out[(row0 + rr) * 256 + n] = f2bf(y);
    }
}

// ---------------------------------------------------------------------------
// Linear head: [rows,256] @ [256,dim(pad16)] + bl, then mask / sigmoid.
// 4 waves per block, each wave one 16-row tile.
// ---------------------------------------------------------------------------
__global__ __launch_bounds__(128)
void linear_head(const unsigned short* __restrict__ h,
                 const unsigned short* __restrict__ wlp,
                 const float* __restrict__ bl, int dim,
                 const unsigned char* __restrict__ mask,
                 float* __restrict__ out, int do_sigmoid) {
    int wid = threadIdx.x >> 5, L = threadIdx.x & 31;
    long tile = (long)blockIdx.x * 4 + wid;        // 1024 tiles
    long row0 = tile * 16;
    int n = L & 15, grp = L >> 4;
    float bcol = (n < dim) ? bl[n] : 0.f;
    v8f acc;
#pragma unroll
    for (int j = 0; j < 8; ++j) acc[j] = bcol;
#pragma unroll
    for (int kc = 0; kc < 8; ++kc) {
        union { v16bf v; uint4 q[2]; } A, Bf;
        const unsigned short* ap = h + (row0 + (L & 15)) * 256 + kc * 32 + grp * 8;
        A.q[0] = *(const uint4*)(ap);
        A.q[1] = *(const uint4*)(ap + 16);
        const unsigned short* bp = wlp + (size_t)kc * 512 + L * 16;
        Bf.q[0] = *(const uint4*)(bp);
        Bf.q[1] = *(const uint4*)(bp + 8);
        acc = __builtin_amdgcn_wmma_f32_16x16x32_bf16(
            false, A.v, false, Bf.v, (short)0, acc, false, false);
    }
    if (n < dim) {
#pragma unroll
        for (int j = 0; j < 8; ++j) {
            long r = row0 + j + 8 * grp;
            float v = acc[j];
            if (mask[r]) v = 0.f;
            if (do_sigmoid) v = 1.f / (1.f + __expf(-v));
            if (dim == 1) out[r] = v;
            else          out[r * dim + n] = v;
        }
    }
}

// ---------------------------------------------------------------------------
// x init + embedding adds (keep f32 running x and bf16 mirror in sync)
// ---------------------------------------------------------------------------
__global__ void init_copy(const float* __restrict__ x, float* __restrict__ xc,
                          unsigned short* __restrict__ xb) {
    long i = (long)blockIdx.x * 256 + threadIdx.x;
    float v = x[i];
    xc[i] = v;
    xb[i] = f2bf(v);
}

__global__ void ed_add(const float* __restrict__ edt, const float* __restrict__ emb,
                       float* __restrict__ xc, unsigned short* __restrict__ xb) {
    __shared__ int idx[12];
    long row = blockIdx.x;
    int t = threadIdx.x;
    if (t < 12) {
        float v = edt[row * 12 + t];
        int id = (int)ceilf(v * 254.f);          // linspace(0,1,255): d = 1/254
        id = id < 0 ? 0 : (id > 255 ? 255 : id);
        idx[t] = id;
    }
    __syncthreads();
    float s = 0.f;
#pragma unroll
    for (int j = 0; j < 12; ++j) s += emb[((size_t)j * NB + idx[j]) * HH + t];
    long i = row * 256 + t;
    float nv = xc[i] + s * (1.f / 12.f);
    xc[i] = nv;
    xb[i] = f2bf(nv);
}

__global__ void scalar_add(const float* __restrict__ tgt, const float* __restrict__ emb,
                           float lo, float invd,
                           float* __restrict__ xc, unsigned short* __restrict__ xb) {
    long row = blockIdx.x;
    int t = threadIdx.x;
    float v = tgt[row];
    int id = (int)ceilf((v - lo) * invd);
    id = id < 0 ? 0 : (id > 255 ? 255 : id);
    long i = row * 256 + t;
    float nv = xc[i] + emb[(size_t)id * HH + t];
    xc[i] = nv;
    xb[i] = f2bf(nv);
}

// ---------------------------------------------------------------------------
// Length regulator
// ---------------------------------------------------------------------------
__global__ void cumsum_k(const int* __restrict__ dur, int* __restrict__ cum,
                         const int* __restrict__ maxlen, float* __restrict__ mel) {
    __shared__ int sc[SS];
    int b = blockIdx.x, t = threadIdx.x;
    sc[t] = dur[b * SS + t];
    __syncthreads();
    for (int off = 1; off < SS; off <<= 1) {
        int v = (t >= off) ? sc[t - off] : 0;
        __syncthreads();
        sc[t] += v;
        __syncthreads();
    }
    cum[b * SS + t] = sc[t];
    if (t == 0) {
        int tot = sc[SS - 1];
        int ml = maxlen[0];
        mel[b] = (float)(tot < ml ? tot : ml);
    }
}

__global__ void gather_k(const float* __restrict__ xc, const int* __restrict__ cum,
                         float* __restrict__ out) {
    __shared__ int sp, sv;
    long bt = blockIdx.x;
    int b = (int)(bt >> 11);                 // TMAX = 2048
    int t = (int)(bt & (TMAXX - 1));
    int h = threadIdx.x;
    if (h == 0) {
        const int* c = cum + (size_t)b * SS;
        int lo = 0, hi = SS;                 // searchsorted side='right'
        while (lo < hi) { int mid = (lo + hi) >> 1; if (c[mid] <= t) lo = mid + 1; else hi = mid; }
        sv = (t < c[SS - 1]) ? 1 : 0;
        sp = lo > (SS - 1) ? (SS - 1) : lo;
    }
    __syncthreads();
    float v = sv ? xc[((size_t)b * SS + sp) * HH + h] : 0.f;
    out[bt * HH + h] = v;
}

// ---------------------------------------------------------------------------
extern "C" void kernel_launch(void* const* d_in, const int* in_sizes, int n_in,
                              void* d_out, int out_size, void* d_ws, size_t ws_size,
                              hipStream_t stream) {
    (void)in_sizes; (void)n_in; (void)out_size; (void)ws_size;

    const float*         x      = (const float*)d_in[0];
    const unsigned char* mask   = (const unsigned char*)d_in[1];
    const float*         edt    = (const float*)d_in[2];
    const float*         ptt    = (const float*)d_in[3];
    const float*         ent    = (const float*)d_in[4];
    const int*           dur    = (const int*)d_in[5];
    const int*           maxlen = (const int*)d_in[6];
    // params: dur(7..16), pitch(17..26), energy(27..36), ed(37..46)
    const float* pitch_emb  = (const float*)d_in[47];
    const float* energy_emb = (const float*)d_in[48];
    const float* ed_emb     = (const float*)d_in[49];

    float* outf = (float*)d_out;
    const long ROWS = (long)BB * SS;                       // 16384
    const long o_out    = 0;
    const long o_mel    = (long)BB * TMAXX * HH;           // 16,777,216
    const long o_logd   = o_mel + BB;
    const long o_pitch  = o_logd + ROWS;
    const long o_energy = o_pitch + ROWS;
    const long o_ed     = o_energy + ROWS;

    // workspace layout (bytes)
    char* ws = (char*)d_ws;
    float*          xc  = (float*)ws;                                  // 16 MB
    unsigned short* xb  = (unsigned short*)(ws + 16777216);            // 8 MB
    unsigned short* h1  = (unsigned short*)(ws + 25165824);            // 8 MB
    unsigned short* h2  = (unsigned short*)(ws + 33554432);            // 8 MB
    unsigned short* wpk = (unsigned short*)(ws + 41943040);            // packed
    int*            cum = (int*)(ws + 45121536);

    // packed weights: per predictor 196608 (w1) + 196608 (w2) + 4096 (wl) ushorts
    const size_t PSTRIDE = 397312;
    unsigned short* w1p[4], *w2p[4], *wlp[4];
    const float *w1[4], *b1[4], *g1[4], *be1[4], *w2[4], *b2[4], *g2[4], *be2[4], *wl[4], *bl[4];
    for (int p = 0; p < 4; ++p) {
        int base = 7 + p * 10;
        w1[p]  = (const float*)d_in[base + 0];
        b1[p]  = (const float*)d_in[base + 1];
        g1[p]  = (const float*)d_in[base + 2];
        be1[p] = (const float*)d_in[base + 3];
        w2[p]  = (const float*)d_in[base + 4];
        b2[p]  = (const float*)d_in[base + 5];
        g2[p]  = (const float*)d_in[base + 6];
        be2[p] = (const float*)d_in[base + 7];
        wl[p]  = (const float*)d_in[base + 8];
        bl[p]  = (const float*)d_in[base + 9];
        w1p[p] = wpk + p * PSTRIDE;
        w2p[p] = wpk + p * PSTRIDE + 196608;
        wlp[p] = wpk + p * PSTRIDE + 393216;
    }
    const int pdim[4] = {1, 1, 1, 12};   // dur, pitch, energy, ed

    // --- pack weights to bf16 WMMA fragments ---
    for (int p = 0; p < 4; ++p) {
        pack_conv_w<<<48, 256, 0, stream>>>(w1[p], w1p[p]);
        pack_conv_w<<<48, 256, 0, stream>>>(w2[p], w2p[p]);
        pack_lin_w<<<1, 256, 0, stream>>>(wl[p], pdim[p], wlp[p]);
    }

    // --- running x (f32) + bf16 mirror ---
    init_copy<<<16384, 256, 0, stream>>>(x, xc, xb);

    auto run_pred = [&](int p, float* outp, int sig) {
        conv_ln<<<1024, 512, 0, stream>>>(xb, w1p[p], b1[p], g1[p], be1[p], h1);
        conv_ln<<<1024, 512, 0, stream>>>(h1, w2p[p], b2[p], g2[p], be2[p], h2);
        linear_head<<<256, 128, 0, stream>>>(h2, wlp[p], bl[p], pdim[p], mask, outp, sig);
    };

    // prosody order: ed, duration, pitch, energy
    run_pred(3, outf + o_ed, 1);                                   // ed (sigmoid, dim 12)
    ed_add<<<(int)ROWS, 256, 0, stream>>>(edt, ed_emb, xc, xb);
    run_pred(0, outf + o_logd, 0);                                 // duration
    run_pred(1, outf + o_pitch, 0);                                // pitch
    scalar_add<<<(int)ROWS, 256, 0, stream>>>(ptt, pitch_emb, -3.f, 254.f / 6.f, xc, xb);
    run_pred(2, outf + o_energy, 0);                               // energy
    scalar_add<<<(int)ROWS, 256, 0, stream>>>(ent, energy_emb, -3.f, 254.f / 6.f, xc, xb);

    // --- length regulation ---
    cumsum_k<<<BB, SS, 0, stream>>>(dur, cum, maxlen, outf + o_mel);
    gather_k<<<BB * TMAXX, 256, 0, stream>>>(xc, cum, outf + o_out);
}
